// MHMDA_47287589929187
// MI455X (gfx1250) — compile-verified, matrix operand order checked by reading, not run on hardware
//
#include <hip/hip_runtime.h>
#include <hip/hip_bf16.h>
#include <cmath>

// ---------------- problem constants ----------------
#define B_TOTAL 8192
#define FDIM    128
#define LROWS   34          // 2 + 32 supernodes
#define LPAD    48          // padded to 3 M-tiles; pad rows are exact zeros
#define LAYERS  3
#define HEADS   4

// ---------------- mapping constants ----------------
#define NB      2           // batch elements per workgroup
#define BLOCK   256         // 8 wave32s
#define NWAVES  (BLOCK/32)
#define HLD     128         // f32 residual h stride
#define GLD     48          // f32 Gram/cos stride (padded: stores unconditional)
#define TBLD    136         // bf16 t stride   (>=128, rows 272B: 16B aligned)
#define HTLD    72          // bf16 h^T stride (>=64,  rows 144B: 16B aligned)
#define PLLD    72          // bf16 pL stride  (>=64)
#define PLK     64          // pL K padded 34 -> 64

// pre-swizzled weight regions in d_ws (bf16 element offsets)
#define WSE_GCN   0         // 3 x [128x128], KS=4
#define WSE_ATTM  49152     // 4 x [128x128], KS=4
#define WSE_ATTD  114688    // 4 x [128x128], KS=4
#define WSE_MM    180224    // [512x128], KS=16
#define WSE_MD    245760    // [512x128], KS=16
#define WSE_TOTAL 311296    // elems (x2 bytes = 622592 B)

typedef __bf16 v16bf __attribute__((ext_vector_type(16)));
typedef __bf16 v8bf  __attribute__((ext_vector_type(8)));
typedef float  f32x8 __attribute__((ext_vector_type(8)));

static __device__ inline f32x8 wmma_bf16(v16bf a, v16bf b, f32x8 c) {
  return __builtin_amdgcn_wmma_f32_16x16x32_bf16(
      false, a, false, b, (short)0, c, false, false);
}

// contiguous bf16 fragment: e[0..7] at row+koff, e[8..15] at row+koff+16
static __device__ inline v16bf lds_frag(const __bf16* row, int koff) {
  v8bf lo = *(const v8bf*)(row + koff);
  v8bf hi = *(const v8bf*)(row + koff + 16);
  return __builtin_shufflevector(lo, hi, 0,1,2,3,4,5,6,7,8,9,10,11,12,13,14,15);
}

// pre-swizzled B fragment: one coalesced 32B load per lane
static __device__ inline v16bf wsfrag(const __bf16* region, int nt, int ks,
                                      int KS, int lane) {
  return *(const v16bf*)(region + (((nt * KS) + ks) << 9) + (lane << 4));
}

// fallback: B fragment gathered from f32 row-major W[k][n]
static __device__ inline v16bf bfrag_f32(const float* __restrict__ base,
                                         int ld, int n, int k0, int kbase) {
  v16bf v;
#pragma unroll
  for (int j = 0; j < 8; ++j) {
    v[j]     = (__bf16)base[(k0 + kbase + j) * ld + n];
    v[j + 8] = (__bf16)base[(k0 + kbase + 16 + j) * ld + n];
  }
  return v;
}

// ---------------- weight pre-swizzle kernel ----------------
static __device__ inline float fetch_swz(const float* __restrict__ W,
                                         int rem, int KS) {
  int block = rem >> 9;          // 512 elems per 16x16x32 B-fragment block
  int inb   = rem & 511;
  int lane  = inb >> 4;
  int j16   = inb & 15;
  int nt    = block / KS;
  int ks    = block % KS;
  int kb    = (lane >> 4) * 8;
  int lmn   = lane & 15;
  int k     = ks * 32 + kb + (j16 < 8 ? j16 : (j16 + 8)); // j16>=8 -> +16+(j16-8)
  return W[k * FDIM + (nt * 16 + lmn)];
}

__global__ __launch_bounds__(BLOCK)
void mhmda_prep(const float* __restrict__ gcnW,
                const float* __restrict__ aWm, const float* __restrict__ aWd,
                const float* __restrict__ mWm, const float* __restrict__ mWd,
                __bf16* __restrict__ ws) {
  for (int idx = blockIdx.x * BLOCK + threadIdx.x; idx < WSE_TOTAL;
       idx += gridDim.x * BLOCK) {
    float v;
    if (idx < WSE_ATTM) {
      int m = idx / 16384, r = idx % 16384;
      v = fetch_swz(gcnW + m * 16384, r, 4);
    } else if (idx < WSE_ATTD) {
      int t = idx - WSE_ATTM; int m = t / 16384, r = t % 16384;
      v = fetch_swz(aWm + m * 16384, r, 4);
    } else if (idx < WSE_MM) {
      int t = idx - WSE_ATTD; int m = t / 16384, r = t % 16384;
      v = fetch_swz(aWd + m * 16384, r, 4);
    } else if (idx < WSE_MD) {
      v = fetch_swz(mWm, idx - WSE_MM, 16);
    } else {
      v = fetch_swz(mWd, idx - WSE_MD, 16);
    }
    ws[idx] = (__bf16)v;
  }
}

// ---------------- fused model kernel ----------------
template <bool SWZ>
__global__ __launch_bounds__(BLOCK)
void mhmda_fused(const float* __restrict__ em,
                 const float* __restrict__ ed,
                 const float* __restrict__ node_emb,
                 const float* __restrict__ gcnW,
                 const float* __restrict__ att_m_w,
                 const float* __restrict__ att_m_W,
                 const float* __restrict__ att_m_b,
                 const float* __restrict__ merge_m_W,
                 const float* __restrict__ merge_m_b,
                 const float* __restrict__ att_d_w,
                 const float* __restrict__ att_d_W,
                 const float* __restrict__ att_d_b,
                 const float* __restrict__ merge_d_W,
                 const float* __restrict__ merge_d_b,
                 const float* __restrict__ fc_W,
                 const float* __restrict__ fc_b,
                 const __bf16* __restrict__ ws,
                 float* __restrict__ out)
{
  __shared__ float                sh_h  [NB][LPAD][HLD];   // residual (f32), pad=0
  __shared__ float                sh_g  [NB][LPAD][GLD];   // Gram / cos (f32)
  __shared__ __align__(16) __bf16 sh_tb [NB][LPAD][TBLD];  // node copy, then t
  __shared__ __align__(16) __bf16 sh_hbT[NB][FDIM][HTLD];  // h^T bf16, K-pad=0
  __shared__ __align__(16) __bf16 sh_pl [NB][LPAD][PLLD];  // pL bf16, pads=0
  __shared__ float  sh_dist[NB][LROWS];
  __shared__ float  sh_dinv[NB][LROWS];
  __shared__ float  sh_mg[NB][LAYERS][FDIM];
  __shared__ float  sh_dg[NB][LAYERS][FDIM];
  __shared__ __bf16 sh_ctxb[NB][HEADS][FDIM];
  __shared__ __bf16 sh_hob [NB][HEADS * FDIM];
  __shared__ float  sh_sc [NB][HEADS][4];
  __shared__ float  sh_merge[2][NB][FDIM];

  const int tid   = threadIdx.x;
  const int lane  = tid & 31;
  const int wave  = tid >> 5;
  const int half  = lane >> 4;
  const int lm    = lane & 15;
  const int kbase = half * 8;
  const int bid   = blockIdx.x;

  // ---- Phase A: node features -> h(f32), tb(bf16), hbT(bf16^T); pads = 0
  for (int i = tid; i < NB * LPAD * FDIM; i += BLOCK) {
    int b = i / (LPAD * FDIM);
    int r = (i / FDIM) % LPAD;
    int f = i % FDIM;
    int gb = bid * NB + b;
    float v = 0.f;
    if (r == 0)          v = em[gb * FDIM + f];
    else if (r == 1)     v = ed[gb * FDIM + f];
    else if (r < LROWS)  v = node_emb[(r - 2) * FDIM + f];
    sh_h[b][r][f]   = v;
    sh_tb[b][r][f]  = (__bf16)v;
    sh_hbT[b][f][r] = (__bf16)v;      // cols 0..47
  }
  for (int i = tid; i < NB * FDIM * (PLK - LPAD); i += BLOCK) {   // cols 48..63
    int b = i / (FDIM * (PLK - LPAD));
    int f = (i / (PLK - LPAD)) % FDIM;
    int z = i % (PLK - LPAD);
    sh_hbT[b][f][LPAD + z] = (__bf16)0.0f;
  }
  __syncthreads();

  // ---- Phase B: Gram = node @ node^T  (pad rows are zero: no clamps/guards)
  for (int t = wave; t < NB * 9; t += NWAVES) {
    int b  = t / 9;
    int mt = (t % 9) / 3;
    int nt = t % 3;
    int ar = mt * 16 + lm;
    int an = nt * 16 + lm;
    f32x8 acc = {0.f,0.f,0.f,0.f,0.f,0.f,0.f,0.f};
#pragma unroll
    for (int ks = 0; ks < 4; ++ks) {
      v16bf av = lds_frag(&sh_tb[b][ar][0], ks * 32 + kbase);
      v16bf bv = lds_frag(&sh_tb[b][an][0], ks * 32 + kbase);
      acc = wmma_bf16(av, bv, acc);
    }
#pragma unroll
    for (int i = 0; i < 8; ++i)
      sh_g[b][mt * 16 + half * 8 + i][nt * 16 + lm] = acc[i];
  }
  __syncthreads();

  // ---- Phase B2: cosine -> leaky -> mask -> D^-1/2 A D^-1/2 -> pL (bf16)
  for (int i = tid; i < NB * LROWS; i += BLOCK) {
    int b = i / LROWS, r = i % LROWS;
    sh_dist[b][r] = sqrtf(sh_g[b][r][r] + 1e-8f);
  }
  __syncthreads();
  for (int i = tid; i < NB * LROWS * LROWS; i += BLOCK) {
    int b = i / (LROWS * LROWS);
    int r = (i / LROWS) % LROWS;
    int c = i % LROWS;
    float cs = sh_g[b][r][c] / (sh_dist[b][r] * sh_dist[b][c] + 1e-8f);
    cs = (cs > 0.f) ? cs : 0.01f * cs;
    if (r == c) cs = 1.0f;
    if ((r == 0 && c == 1) || (r == 1 && c == 0)) cs = 0.0f;
    sh_g[b][r][c] = cs;
  }
  __syncthreads();
  for (int i = tid; i < NB * LROWS; i += BLOCK) {
    int b = i / LROWS, r = i % LROWS;
    float s = 0.f;
    for (int c = 0; c < LROWS; ++c) s += sh_g[b][r][c];
    sh_dinv[b][r] = rsqrtf(s);
  }
  __syncthreads();
  for (int i = tid; i < NB * LPAD * PLK; i += BLOCK) {
    int b = i / (LPAD * PLK);
    int r = (i / PLK) % LPAD;
    int c = i % PLK;
    float v = (r < LROWS && c < LROWS)
                ? sh_dinv[b][r] * sh_g[b][r][c] * sh_dinv[b][c] : 0.f;
    sh_pl[b][r][c] = (__bf16)v;
  }
  __syncthreads();

  // ---- Phase C: residual GCN layers (all stores unconditional) ----
  for (int layer = 0; layer < LAYERS; ++layer) {
    // C1: t = pL @ h.  wave <-> ntile; B frags (h^T) hoisted per batch elem.
    {
      int nt = wave;
      int n  = nt * 16 + lm;
      for (int b = 0; b < NB; ++b) {
        v16bf bw0 = lds_frag(&sh_hbT[b][n][0], kbase);
        v16bf bw1 = lds_frag(&sh_hbT[b][n][0], 32 + kbase);
#pragma unroll
        for (int mt = 0; mt < 3; ++mt) {
          int ar = mt * 16 + lm;
          f32x8 acc = {0.f,0.f,0.f,0.f,0.f,0.f,0.f,0.f};
          acc = wmma_bf16(lds_frag(&sh_pl[b][ar][0], kbase),      bw0, acc);
          acc = wmma_bf16(lds_frag(&sh_pl[b][ar][0], 32 + kbase), bw1, acc);
#pragma unroll
          for (int i = 0; i < 8; ++i)
            sh_tb[b][mt * 16 + half * 8 + i][n] = (__bf16)acc[i];
        }
      }
    }
    __syncthreads();

    // C2: h = relu(t @ W) + h.  Per-batch tiling over padded rows.
    {
      const float* W = gcnW + layer * FDIM * FDIM;
      int nt = wave;
      int n  = nt * 16 + lm;
      v16bf bw[4];
#pragma unroll
      for (int ks = 0; ks < 4; ++ks) {
        if constexpr (SWZ)
          bw[ks] = wsfrag(ws + WSE_GCN + layer * 16384, nt, ks, 4, lane);
        else
          bw[ks] = bfrag_f32(W, FDIM, n, ks * 32, kbase);
      }
      for (int b = 0; b < NB; ++b) {
#pragma unroll
        for (int mt = 0; mt < 3; ++mt) {
          int ar = mt * 16 + lm;
          f32x8 acc = {0.f,0.f,0.f,0.f,0.f,0.f,0.f,0.f};
#pragma unroll
          for (int ks = 0; ks < 4; ++ks)
            acc = wmma_bf16(lds_frag(&sh_tb[b][ar][0], ks * 32 + kbase),
                            bw[ks], acc);
#pragma unroll
          for (int i = 0; i < 8; ++i) {
            int m = mt * 16 + half * 8 + i;
            float v = fmaxf(acc[i], 0.f) + sh_h[b][m][n];
            sh_h[b][m][n]   = v;                  // pad rows stay 0
            sh_hbT[b][n][m] = (__bf16)v;          // pad cols stay 0
          }
        }
      }
    }
    __syncthreads();

    for (int i = tid; i < NB * FDIM; i += BLOCK) {
      int b = i / FDIM, f = i % FDIM;
      sh_mg[b][layer][f] = sh_h[b][0][f];
      sh_dg[b][layer][f] = sh_h[b][1][f];
    }
    __syncthreads();
  }

  // ---- Phase D: multi-head attention, two branches ----
  for (int br = 0; br < 2; ++br) {
    const float* G  = br ? &sh_dg[0][0][0] : &sh_mg[0][0][0];
    const float* aw = br ? att_d_w   : att_m_w;
    const float* aW = br ? att_d_W   : att_m_W;
    const float* ab = br ? att_d_b   : att_m_b;
    const float* mW = br ? merge_d_W : merge_m_W;
    const float* mb = br ? merge_d_b : merge_m_b;
    const __bf16* aWs = ws + (br ? WSE_ATTD : WSE_ATTM);
    const __bf16* mWs = ws + (br ? WSE_MD   : WSE_MM);

    // D1: scores[b][h][l]
    if (tid < NB * HEADS * LAYERS) {
      int b  = tid / (HEADS * LAYERS);
      int h4 = (tid / LAYERS) % HEADS;
      int j  = tid % LAYERS;
      float s = 0.f;
      for (int f = 0; f < FDIM; ++f)
        s += G[(b * LAYERS + j) * FDIM + f] * aw[h4 * FDIM + f];
      sh_sc[b][h4][j] = s;
    }
    __syncthreads();

    // D2: softmax over layers + ctx (store bf16, only WMMA consumes it)
    for (int i = tid; i < NB * HEADS * FDIM; i += BLOCK) {
      int b  = i / (HEADS * FDIM);
      int h4 = (i / FDIM) % HEADS;
      int f  = i % FDIM;
      float s0 = sh_sc[b][h4][0], s1 = sh_sc[b][h4][1], s2 = sh_sc[b][h4][2];
      float mx = fmaxf(s0, fmaxf(s1, s2));
      float e0 = expf(s0 - mx), e1 = expf(s1 - mx), e2 = expf(s2 - mx);
      float inv = 1.f / (e0 + e1 + e2);
      float c = (e0 * G[(b * LAYERS + 0) * FDIM + f]
               + e1 * G[(b * LAYERS + 1) * FDIM + f]
               + e2 * G[(b * LAYERS + 2) * FDIM + f]) * inv;
      sh_ctxb[b][h4][f] = (__bf16)c;
    }
    __syncthreads();

    // D3: per-head linear ho = ctx @ aW[h] + ab[h]
    for (int t = wave; t < HEADS * 8; t += NWAVES) {
      int h4 = t / 8;
      int nt = t % 8;
      int n  = nt * 16 + lm;
      f32x8 acc = {0.f,0.f,0.f,0.f,0.f,0.f,0.f,0.f};
#pragma unroll
      for (int ks = 0; ks < 4; ++ks) {
        v16bf av;
#pragma unroll
        for (int j = 0; j < 8; ++j) {
          av[j]     = (lm < NB) ? sh_ctxb[lm][h4][ks * 32 + kbase + j]      : (__bf16)0.0f;
          av[j + 8] = (lm < NB) ? sh_ctxb[lm][h4][ks * 32 + kbase + 16 + j] : (__bf16)0.0f;
        }
        v16bf bv;
        if constexpr (SWZ) bv = wsfrag(aWs + h4 * 16384, nt, ks, 4, lane);
        else               bv = bfrag_f32(aW + h4 * FDIM * FDIM, FDIM, n, ks * 32, kbase);
        acc = wmma_bf16(av, bv, acc);
      }
      if (half == 0) {   // rows 0..1 live in acc[0..1] of lanes 0..15
        sh_hob[0][h4 * FDIM + n] = (__bf16)(acc[0] + ab[h4 * FDIM + n]);
        sh_hob[1][h4 * FDIM + n] = (__bf16)(acc[1] + ab[h4 * FDIM + n]);
      }
    }
    __syncthreads();

    // D4: merge: [NB,512] @ mW + mb
    for (int t = wave; t < 8; t += NWAVES) {
      int nt = t;
      int n  = nt * 16 + lm;
      f32x8 acc = {0.f,0.f,0.f,0.f,0.f,0.f,0.f,0.f};
#pragma unroll
      for (int ks = 0; ks < 16; ++ks) {
        v16bf av;
#pragma unroll
        for (int j = 0; j < 8; ++j) {
          av[j]     = (lm < NB) ? sh_hob[lm][ks * 32 + kbase + j]      : (__bf16)0.0f;
          av[j + 8] = (lm < NB) ? sh_hob[lm][ks * 32 + kbase + 16 + j] : (__bf16)0.0f;
        }
        v16bf bv;
        if constexpr (SWZ) bv = wsfrag(mWs, nt, ks, 16, lane);
        else               bv = bfrag_f32(mW, FDIM, n, ks * 32, kbase);
        acc = wmma_bf16(av, bv, acc);
      }
      if (half == 0) {
        sh_merge[br][0][n] = acc[0] + mb[n];
        sh_merge[br][1][n] = acc[1] + mb[n];
      }
    }
    __syncthreads();
  }

  // ---- Phase E: sigmoid((mA*dA) @ fc_W + fc_b) ----
  if (tid < NB) {
    float z = fc_b[0];
    for (int f = 0; f < FDIM; ++f)
      z += sh_merge[0][tid][f] * sh_merge[1][tid][f] * fc_W[f];
    out[bid * NB + tid] = 1.f / (1.f + expf(-z));
  }
}

extern "C" void kernel_launch(void* const* d_in, const int* in_sizes, int n_in,
                              void* d_out, int out_size, void* d_ws, size_t ws_size,
                              hipStream_t stream) {
  (void)in_sizes; (void)n_in; (void)out_size;
  const float* em    = (const float*)d_in[0];
  const float* ed    = (const float*)d_in[1];
  const float* nemb  = (const float*)d_in[2];
  const float* gcnW  = (const float*)d_in[3];
  const float* amw   = (const float*)d_in[4];
  const float* amW   = (const float*)d_in[5];
  const float* amb   = (const float*)d_in[6];
  const float* mmW   = (const float*)d_in[7];
  const float* mmb   = (const float*)d_in[8];
  const float* adw   = (const float*)d_in[9];
  const float* adW   = (const float*)d_in[10];
  const float* adb   = (const float*)d_in[11];
  const float* mdW   = (const float*)d_in[12];
  const float* mdb   = (const float*)d_in[13];
  const float* fcW   = (const float*)d_in[14];
  const float* fcb   = (const float*)d_in[15];
  float* out = (float*)d_out;

  dim3 grid(B_TOTAL / NB), block(BLOCK);
  if (ws_size >= (size_t)WSE_TOTAL * sizeof(__bf16)) {
    __bf16* ws = (__bf16*)d_ws;
    mhmda_prep<<<dim3(512), block, 0, stream>>>(gcnW, amW, adW, mmW, mdW, ws);
    mhmda_fused<true><<<grid, block, 0, stream>>>(
        em, ed, nemb, gcnW, amw, amW, amb, mmW, mmb,
        adw, adW, adb, mdW, mdb, fcW, fcb, ws, out);
  } else {
    mhmda_fused<false><<<grid, block, 0, stream>>>(
        em, ed, nemb, gcnW, amw, amW, amb, mmW, mmb,
        adw, adW, adb, mdW, mdb, fcW, fcb, nullptr, out);
  }
}
